// SlidingWindowAttention_23570780520559
// MI455X (gfx1250) — compile-verified, matrix operand order checked by reading, not run on hardware
//
#include <hip/hip_runtime.h>

typedef __attribute__((ext_vector_type(16))) _Float16 v16h;
typedef __attribute__((ext_vector_type(8)))  float    v8f;

#define B_  2
#define H_  16
#define N_  2048
#define D_  64
#define W_  512
#define WAVES_PER_BLOCK 8
#define SP_HALVES (16*32)      // one 16x32 f16 P tile
#define ROWS_PER_WAVE 32       // two 16-row M-tiles per wave

#define WMMA_F16(A,B,C) __builtin_amdgcn_wmma_f32_16x16x32_f16(false, A, false, B, (short)0, C, false, false)

// Build a 16-half WMMA operand from an f32 row in global memory.
// CDNA5 16-bit operand layout: slots v=0..3 hold K = base+0..7 (pairs/VGPR),
// slots v=4..7 hold K = base+16..23; base already includes kk + 8*laneHi.
// f32 pairs -> v_cvt_pk_f16_f32 packs VGPRs with no extra moves.
__device__ __forceinline__ v16h ld_op_f32row(const float* p, float scale) {
    float4 a = ((const float4*)p)[0];
    float4 b = ((const float4*)p)[1];
    float4 c = ((const float4*)(p + 16))[0];
    float4 d = ((const float4*)(p + 16))[1];
    v16h r;
    r[0]  = (_Float16)(a.x * scale); r[1]  = (_Float16)(a.y * scale);
    r[2]  = (_Float16)(a.z * scale); r[3]  = (_Float16)(a.w * scale);
    r[4]  = (_Float16)(b.x * scale); r[5]  = (_Float16)(b.y * scale);
    r[6]  = (_Float16)(b.z * scale); r[7]  = (_Float16)(b.w * scale);
    r[8]  = (_Float16)(c.x * scale); r[9]  = (_Float16)(c.y * scale);
    r[10] = (_Float16)(c.z * scale); r[11] = (_Float16)(c.w * scale);
    r[12] = (_Float16)(d.x * scale); r[13] = (_Float16)(d.y * scale);
    r[14] = (_Float16)(d.z * scale); r[15] = (_Float16)(d.w * scale);
    return r;
}

// Column gather (stride f32 elements) for the V B-operand: K runs down rows.
// p already includes the (kb + 8*laneHi) row base and the feature column.
__device__ __forceinline__ v16h ld_op_f32col(const float* p, int stride) {
    v16h r;
#pragma unroll
    for (int vv = 0; vv < 8; ++vv) {
        int k0 = ((vv >> 2) << 4) + ((vv & 3) << 1);
        float x = p[(size_t)k0 * stride];
        float y = p[(size_t)(k0 + 1) * stride];
        r[2*vv]   = (_Float16)x;
        r[2*vv+1] = (_Float16)y;
    }
    return r;
}

// Aligned 16B LDS loads for the P operand (row-major 16x32 f16 tile).
__device__ __forceinline__ v16h ld_op_lds(const _Float16* p) {
    union { uint4 u[2]; v16h h; } r;
    r.u[0] = *(const uint4*)(p);
    r.u[1] = *(const uint4*)(p + 16);
    return r.h;
}

__global__ __launch_bounds__(32 * WAVES_PER_BLOCK)
void swa_wmma_f16_kernel(const float* __restrict__ q,
                         const float* __restrict__ k,
                         const float* __restrict__ v,
                         float* __restrict__ out) {
    __shared__ __align__(16) _Float16 sPall[WAVES_PER_BLOCK * 2 * SP_HALVES];

    const int lane   = threadIdx.x & 31;
    const int wave   = threadIdx.x >> 5;
    const int laneHi = lane >> 4;
    const int laneN  = lane & 15;
    const int hi8    = laneHi << 3;

    _Float16* sP0 = sPall + wave * 2 * SP_HALVES;
    _Float16* sP1 = sP0 + SP_HALVES;

    const int tile  = blockIdx.x * WAVES_PER_BLOCK + wave;   // 0..2047
    const int bh    = tile >> 6;                             // 64 tiles per (b,h)
    const int qBase = (tile & 63) << 5;                      // first of 32 query rows

    const float NEG_INF = -__builtin_inff();
    const float qscale  = 0.125f * 1.44269504088896340736f;  // 1/sqrt(D) * log2(e)

    // constant all-ones B operand: rowsum(P) = P x ones via WMMA
    v16h onesB;
#pragma unroll
    for (int j = 0; j < 16; ++j) onesB[j] = (_Float16)1.0f;
    const v8f zero8 = (v8f){0.f,0.f,0.f,0.f,0.f,0.f,0.f,0.f};

    // ---- loop-invariant Q A-operands: [mt][kk-half] ----
    v16h aQ[2][2];
#pragma unroll
    for (int mt = 0; mt < 2; ++mt) {
        const float* qrow = q + ((size_t)bh * N_ + (qBase + (mt << 4) + laneN)) * D_ + hi8;
        aQ[mt][0] = ld_op_f32row(qrow +  0, qscale);
        aQ[mt][1] = ld_op_f32row(qrow + 32, qscale);
    }

    // per-row softmax state; stat [mt][i] is row (qBase + 16*mt + i + 8*laneHi)
    float mrun[2][8], lrun[2][8], alpha[2][8];
    v8f   oacc[2][4];
#pragma unroll
    for (int mt = 0; mt < 2; ++mt) {
#pragma unroll
        for (int i = 0; i < 8; ++i) { mrun[mt][i] = NEG_INF; lrun[mt][i] = 0.f; }
#pragma unroll
        for (int t = 0; t < 4; ++t) oacc[mt][t] = zero8;
    }

    const int lo     = (qBase >= W_) ? (qBase - W_ + 1) : 0;
    const int kb0    = (lo >> 5) << 5;
    const int kbLast = qBase;                    // chunk [qBase, qBase+31] = diagonal

    const float* kbh = k + (size_t)bh * N_ * D_;
    const float* vbh = v + (size_t)bh * N_ * D_;

    for (int kb = kb0; kb <= kbLast; kb += 32) {
        if (kb + 32 <= kbLast) {
            __builtin_prefetch(kbh + (size_t)(kb + 32 + lane) * D_, 0, 3);
            __builtin_prefetch(vbh + (size_t)(kb + 32 + lane) * D_, 0, 3);
        }

        // ---- S = (Q*scale) K^T : 2 M-tiles x 2 N-halves, K=64 in 2 steps ----
        const float* kr0 = kbh + (size_t)(kb + laneN) * D_ + hi8;        // keys kb+0..15
        const float* kr1 = kbh + (size_t)(kb + 16 + laneN) * D_ + hi8;   // keys kb+16..31
        v8f s[2][2] = {{zero8, zero8}, {zero8, zero8}};                  // [mt][nhalf]
#pragma unroll
        for (int kkI = 0; kkI < 2; ++kkI) {
            v16h bK0 = ld_op_f32row(kr0 + kkI * 32, 1.0f);
            v16h bK1 = ld_op_f32row(kr1 + kkI * 32, 1.0f);
            s[0][0] = WMMA_F16(aQ[0][kkI], bK0, s[0][0]);   // each bK feeds 2 M-tiles
            s[1][0] = WMMA_F16(aQ[1][kkI], bK0, s[1][0]);
            s[0][1] = WMMA_F16(aQ[0][kkI], bK1, s[0][1]);
            s[1][1] = WMMA_F16(aQ[1][kkI], bK1, s[1][1]);
        }

        // interior chunks are wave-uniformly unmasked
        const bool needMask = (kb + 31 > qBase) || (kb < qBase - (W_ - 32));

        if (needMask) {
#pragma unroll
            for (int mt = 0; mt < 2; ++mt) {
                _Float16* sP = mt ? sP1 : sP0;
                float rm[8];
#pragma unroll
                for (int i = 0; i < 8; ++i) {
                    int row = qBase + (mt << 4) + i + hi8;
                    int c0  = kb + laneN;
                    int c1  = c0 + 16;
                    float v0 = ((c0 <= row) && (c0 + (W_ - 1) >= row)) ? s[mt][0][i] : NEG_INF;
                    float v1 = ((c1 <= row) && (c1 + (W_ - 1) >= row)) ? s[mt][1][i] : NEG_INF;
                    s[mt][0][i] = v0; s[mt][1][i] = v1;
                    rm[i] = fmaxf(v0, v1);
                }
#pragma unroll
                for (int msk = 1; msk < 16; msk <<= 1)
#pragma unroll
                    for (int i = 0; i < 8; ++i) rm[i] = fmaxf(rm[i], __shfl_xor(rm[i], msk, 32));
#pragma unroll
                for (int i = 0; i < 8; ++i) {
                    float mnew = fmaxf(mrun[mt][i], rm[i]);
                    float a, p0, p1;
                    if (mnew == NEG_INF) { a = 1.f; p0 = 0.f; p1 = 0.f; }
                    else {
                        a  = __builtin_exp2f(mrun[mt][i] - mnew);
                        p0 = __builtin_exp2f(s[mt][0][i] - mnew);
                        p1 = __builtin_exp2f(s[mt][1][i] - mnew);
                    }
                    mrun[mt][i]  = mnew;
                    alpha[mt][i] = a;
                    int m = i + hi8;
                    sP[m * 32 + laneN]      = (_Float16)p0;
                    sP[m * 32 + 16 + laneN] = (_Float16)p1;
                }
            }
        } else {
#pragma unroll
            for (int mt = 0; mt < 2; ++mt) {
                _Float16* sP = mt ? sP1 : sP0;
                float rm[8];
#pragma unroll
                for (int i = 0; i < 8; ++i) rm[i] = fmaxf(s[mt][0][i], s[mt][1][i]);
#pragma unroll
                for (int msk = 1; msk < 16; msk <<= 1)
#pragma unroll
                    for (int i = 0; i < 8; ++i) rm[i] = fmaxf(rm[i], __shfl_xor(rm[i], msk, 32));
#pragma unroll
                for (int i = 0; i < 8; ++i) {
                    float mnew = fmaxf(mrun[mt][i], rm[i]);
                    alpha[mt][i] = __builtin_exp2f(mrun[mt][i] - mnew);  // exp2(-inf)=0 ok
                    float p0 = __builtin_exp2f(s[mt][0][i] - mnew);
                    float p1 = __builtin_exp2f(s[mt][1][i] - mnew);
                    mrun[mt][i] = mnew;
                    int m = i + hi8;
                    sP[m * 32 + laneN]      = (_Float16)p0;
                    sP[m * 32 + 16 + laneN] = (_Float16)p1;
                }
            }
        }

        // ---- P operands back from LDS (C-layout -> A-layout) ----
        v16h aP0 = ld_op_lds(sP0 + laneN * 32 + hi8);
        v16h aP1 = ld_op_lds(sP1 + laneN * 32 + hi8);

        // row sums of P via WMMA against ones: lands per-row in C-layout slots
        v8f rs0 = WMMA_F16(aP0, onesB, zero8);
        v8f rs1 = WMMA_F16(aP1, onesB, zero8);
#pragma unroll
        for (int i = 0; i < 8; ++i) {
            lrun[0][i] = lrun[0][i] * alpha[0][i] + rs0[i];
            lrun[1][i] = lrun[1][i] * alpha[1][i] + rs1[i];
#pragma unroll
            for (int t = 0; t < 4; ++t) {
                oacc[0][t][i] *= alpha[0][i];
                oacc[1][t][i] *= alpha[1][i];
            }
        }

        // ---- O += P (16x32) * V (32x64): each bV feeds both M-tiles ----
        const float* vkb = vbh + (size_t)(kb + hi8) * D_;
#pragma unroll
        for (int t = 0; t < 4; ++t) {
            v16h bV = ld_op_f32col(vkb + t * 16 + laneN, D_);
            oacc[0][t] = WMMA_F16(aP0, bV, oacc[0][t]);
            oacc[1][t] = WMMA_F16(aP1, bV, oacc[1][t]);
        }
    }

    // ---- finalize: divide by l, store fp32 ----
#pragma unroll
    for (int mt = 0; mt < 2; ++mt)
#pragma unroll
        for (int i = 0; i < 8; ++i) {
            float inv = (lrun[mt][i] > 0.f) ? (1.f / lrun[mt][i]) : 0.f;
            int row = qBase + (mt << 4) + i + hi8;
            float* orow = out + ((size_t)bh * N_ + row) * D_;
#pragma unroll
            for (int t = 0; t < 4; ++t) orow[t * 16 + laneN] = oacc[mt][t][i] * inv;
        }
}

extern "C" void kernel_launch(void* const* d_in, const int* in_sizes, int n_in,
                              void* d_out, int out_size, void* d_ws, size_t ws_size,
                              hipStream_t stream) {
    (void)in_sizes; (void)n_in; (void)d_ws; (void)ws_size; (void)out_size;
    const float* q = (const float*)d_in[0];
    const float* k = (const float*)d_in[1];
    const float* v = (const float*)d_in[2];
    float* out = (float*)d_out;

    const int tiles  = B_ * H_ * (N_ / ROWS_PER_WAVE);   // 2048 query tiles
    const int blocks = tiles / WAVES_PER_BLOCK;          // 256 blocks of 8 wave32
    hipLaunchKernelGGL(swa_wmma_f16_kernel,
                       dim3(blocks), dim3(32 * WAVES_PER_BLOCK), 0, stream,
                       q, k, v, out);
}
